// MLA_48249662603688
// MI455X (gfx1250) — matrix operand order repaired and checked by reading, not compiled, not run on hardware
//
#include <hip/hip_runtime.h>
#include <hip/hip_bf16.h>
#include <stdint.h>

typedef __bf16 bf16_t;
typedef __attribute__((ext_vector_type(16))) __bf16 v16bf;
typedef __attribute__((ext_vector_type(8)))  float  v8f;
typedef __attribute__((ext_vector_type(4)))  unsigned int u32x4;
typedef __attribute__((ext_vector_type(8)))  int i32x8;
typedef __attribute__((ext_vector_type(4)))  int i32x4;

union V16 { v16bf v; uint4 u[2]; };
union U4B { uint4 u; bf16_t e[8]; };

#define S_    2048
#define E_    2048
#define H_    16
#define QL_   1536
#define KL_   512
#define DN_   128
#define DR_   64
#define DV_   128
#define DQ_   192            // DN + DR
#define HDQ_  (H_ * DQ_)     // 3072
#define HDNV_ (H_ * (DN_ + DV_)) // 4096
#define HDV_  (H_ * DV_)     // 2048

#if defined(__has_builtin)
#  if __has_builtin(__builtin_amdgcn_tensor_load_to_lds)
#    define HAVE_TDM 1
#  endif
#endif
#ifndef HAVE_TDM
#  define HAVE_TDM 0
#endif

__device__ __forceinline__ v8f wmma_bf16(v16bf a, v16bf b, v8f c) {
  // (neg_a, A, neg_b, B, c_mod, C, reuse_a, reuse_b)
  return __builtin_amdgcn_wmma_f32_16x16x32_bf16(false, a, false, b, (short)0, c, false, false);
}

#if HAVE_TDM
// Issue a 2-D TDM load of a (tile_d1 x tile_d0) bf16 tile into LDS (row-major,
// contiguous). Dims/strides in elements (data_size = 2 bytes). D# bit layout per
// CDNA5 ISA 8.3/8.4 (group0: count/lds/global/type; group1: dsize/dims/stride).
// This toolchain uses the 6-arg builtin: (g0 u32x4, g1 i32x8, g2 i32x4, g3 i32x4,
// g4 i32x8, cpol i32).
__device__ __forceinline__ void tdm_load_tile_2d(const bf16_t* gptr, unsigned lds_off,
                                                 unsigned tensor_d0, unsigned tensor_d1,
                                                 unsigned tile_d0, unsigned tile_d1,
                                                 unsigned long long stride0) {
  unsigned long long ga = (unsigned long long)(uintptr_t)gptr;
  u32x4 g0;
  g0[0] = 1u;                                            // count=1, user descriptor
  g0[1] = lds_off;                                       // lds_addr (bytes)
  g0[2] = (unsigned)(ga & 0xFFFFFFFFu);                  // global_addr[31:0]
  g0[3] = (unsigned)((ga >> 32) & 0x01FFFFFFu) | (2u << 30); // addr[56:32] | type=2
  i32x8 g1;
  g1[0] = (int)(1u << 16);                               // data_size=1 (2 bytes)
  g1[1] = (int)((tensor_d0 & 0xFFFFu) << 16);            // tensor_dim0[15:0]
  g1[2] = (int)((tensor_d0 >> 16) | ((tensor_d1 & 0xFFFFu) << 16));
  g1[3] = (int)((tensor_d1 >> 16) | (tile_d0 << 16));    // tensor_dim1[31:16] | tile_dim0
  g1[4] = (int)(tile_d1 & 0xFFFFu);                      // tile_dim1 (tile_dim2 = 0)
  g1[5] = (int)(stride0 & 0xFFFFFFFFull);                // tensor_dim0_stride[31:0]
  g1[6] = (int)((stride0 >> 32) & 0xFFFFull);            // stride[47:32]
  g1[7] = 0;
  i32x4 z4 = {0, 0, 0, 0};
  i32x8 z8 = {0, 0, 0, 0, 0, 0, 0, 0};
  __builtin_amdgcn_tensor_load_to_lds(g0, g1, z4, z4, z8, 0);
}
#endif

// ---------------------------------------------------------------- convert
__global__ __launch_bounds__(256) void k_f32_to_bf16(const float* __restrict__ in,
                                                     bf16_t* __restrict__ out, int n) {
  int i = blockIdx.x * blockDim.x + threadIdx.x;
  int stride = gridDim.x * blockDim.x;
  for (; i < n; i += stride) out[i] = (bf16_t)in[i];
}

// ---------------------------------------------------------------- rmsnorm
__global__ __launch_bounds__(256) void k_rmsnorm(const float* __restrict__ X,
                                                 const float* __restrict__ W,
                                                 bf16_t* __restrict__ Y, int L, int ldx) {
  __shared__ float red[8];
  __shared__ float rn_s;
  const float* x = X + (size_t)blockIdx.x * ldx;
  float ss = 0.f;
  for (int i = threadIdx.x; i < L; i += 256) { float v = x[i]; ss += v * v; }
#pragma unroll
  for (int m = 16; m >= 1; m >>= 1) ss += __shfl_xor(ss, m, 32);
  if ((threadIdx.x & 31) == 0) red[threadIdx.x >> 5] = ss;
  __syncthreads();
  if (threadIdx.x == 0) {
    float t = 0.f;
#pragma unroll
    for (int i = 0; i < 8; ++i) t += red[i];
    rn_s = rsqrtf(t / (float)L + 1e-6f);
  }
  __syncthreads();
  float rn = rn_s;
  bf16_t* y = Y + (size_t)blockIdx.x * L;
  for (int i = threadIdx.x; i < L; i += 256) y[i] = (bf16_t)(x[i] * rn * W[i]);
}

// ---------------------------------------------------------------- bf16 WMMA GEMM
// C(f32, MxN) = A(bf16, MxK, rm) x B(bf16, KxN, rm)
// block tile 128 x (32*NT), 8 waves (4 M x 2 N), wave tile 32 x (16*NT), K-step 32
template <int NT>
__global__ __launch_bounds__(256) void k_gemm(const bf16_t* __restrict__ A,
                                              const bf16_t* __restrict__ B,
                                              float* __restrict__ C,
                                              int M, int N, int K) {
  constexpr int BN = 32 * NT; // 64 or 128
  __shared__ __align__(16) bf16_t As[128][32];
  __shared__ __align__(16) bf16_t BsT[BN][32]; // B tile transposed: [n][k]
  const int tid  = threadIdx.x;
  const int wave = tid >> 5, lane = tid & 31;
  const int half_ = lane >> 4, l16 = lane & 15;
  const int wm = (wave >> 1) * 32;
  const int wn = (wave & 1) * (16 * NT);
  const int bm = blockIdx.x * 128;
  const int bn = blockIdx.y * BN;
  v8f acc[2][NT] = {};
  for (int k0 = 0; k0 < K; k0 += 32) {
#if HAVE_TDM
    // A tile 128x32 via Tensor Data Mover (one instruction, wave 0 only)
    if (wave == 0)
      tdm_load_tile_2d(&A[(size_t)bm * K + k0], (unsigned)(uintptr_t)&As[0][0],
                       (unsigned)K, (unsigned)M, 32u, 128u, (unsigned long long)K);
#else
#pragma unroll
    for (int r = 0; r < 2; ++r) {
      int u = r * 256 + tid;
      int row = u >> 2, c8 = (u & 3) * 8;
      *(uint4*)&As[row][c8] = *(const uint4*)&A[(size_t)(bm + row) * K + k0 + c8];
    }
#endif
    // stage B transposed: 32 x BN bf16, (BN/64) uint4 per thread
#pragma unroll
    for (int r = 0; r < BN / 64; ++r) {
      int u = r * 256 + tid;
      int row = u / (BN / 8), c8 = (u % (BN / 8)) * 8;
      U4B t; t.u = *(const uint4*)&B[(size_t)(k0 + row) * N + bn + c8];
#pragma unroll
      for (int j = 0; j < 8; ++j) BsT[c8 + j][row] = t.e[j];
    }
    if (k0 + 32 < K) __builtin_prefetch(&B[(size_t)(k0 + 32) * N + bn], 0, 1);
#if HAVE_TDM
    if (wave == 0) __builtin_amdgcn_s_wait_tensorcnt(0);
#endif
    __syncthreads();
    // A-frag (16-bit 16x32): lanes 0-15 hold K {0..7,16..23}, lanes 16-31 {8..15,24..31}
    const int kA0 = half_ ? 8 : 0, kA1 = half_ ? 24 : 16;
    // B-frag (16-bit 32x16): lanes 0-15 hold K {0..7,16..23}, lanes 16-31 {8..15,24..31}
    const int kB = half_ * 8;
    V16 a[2];
#pragma unroll
    for (int i = 0; i < 2; ++i) {
      int row = wm + i * 16 + l16;
      a[i].u[0] = *(const uint4*)&As[row][kA0];
      a[i].u[1] = *(const uint4*)&As[row][kA1];
    }
#pragma unroll
    for (int jj = 0; jj < NT; ++jj) {
      V16 b;
      int col = wn + jj * 16 + l16;
      b.u[0] = *(const uint4*)&BsT[col][kB];
      b.u[1] = *(const uint4*)&BsT[col][kB + 16];
#pragma unroll
      for (int i = 0; i < 2; ++i)
        acc[i][jj] = wmma_bf16(a[i].v, b.v, acc[i][jj]);
    }
    __syncthreads();
  }
  // C layout: VGPR p holds row (p + 8*half_), col = l16
#pragma unroll
  for (int i = 0; i < 2; ++i)
#pragma unroll
    for (int jj = 0; jj < NT; ++jj) {
      int row0 = bm + wm + i * 16 + half_ * 8;
      int col  = bn + wn + jj * 16 + l16;
#pragma unroll
      for (int p = 0; p < 8; ++p)
        C[(size_t)(row0 + p) * N + col] = acc[i][jj][p];
    }
}

// ---------------------------------------------------------------- RoPE + head-major repack
__global__ __launch_bounds__(256) void k_rope_pack(const float* __restrict__ q,
                                                   const float* __restrict__ kv,
                                                   const float* __restrict__ ckv,
                                                   bf16_t* __restrict__ qf,
                                                   bf16_t* __restrict__ kf,
                                                   bf16_t* __restrict__ vt) {
  const int s = blockIdx.x;
  __shared__ float cs[32], sn[32], kpe[64];
  if (threadIdx.x < 32) {
    int i = threadIdx.x;
    float inv = __powf(10000.0f, -(2.0f * (float)i) / (float)DR_);
    float ang = (float)s * inv;
    cs[i] = __cosf(ang);
    sn[i] = __sinf(ang);
  }
  __syncthreads();
  if (threadIdx.x < 32) {
    int i = threadIdx.x;
    float x0 = ckv[(size_t)s * (KL_ + DR_) + KL_ + 2 * i];
    float x1 = ckv[(size_t)s * (KL_ + DR_) + KL_ + 2 * i + 1];
    kpe[i]      = x0 * cs[i] - x1 * sn[i];
    kpe[i + 32] = x1 * cs[i] + x0 * sn[i];
  }
  __syncthreads();
  const float* qrow_base = q + (size_t)s * HDQ_;
  for (int e = threadIdx.x; e < HDQ_; e += 256) {
    int hh = e / DQ_, d = e % DQ_;
    const float* qrow = qrow_base + hh * DQ_;
    float val;
    if (d < DN_) {
      val = qrow[d];
    } else {
      int i = d - DN_;
      int f = (i < 32) ? i : (i - 32);
      float x0 = qrow[DN_ + 2 * f], x1 = qrow[DN_ + 2 * f + 1];
      val = (i < 32) ? (x0 * cs[f] - x1 * sn[f]) : (x1 * cs[f] + x0 * sn[f]);
    }
    qf[((size_t)hh * S_ + s) * DQ_ + d] = (bf16_t)val;
  }
  const float* kvrow = kv + (size_t)s * HDNV_;
  for (int e = threadIdx.x; e < HDQ_; e += 256) {
    int hh = e / DQ_, d = e % DQ_;
    float val = (d < DN_) ? kvrow[hh * (DN_ + DV_) + d] : kpe[d - DN_];
    kf[((size_t)hh * S_ + s) * DQ_ + d] = (bf16_t)val;
  }
  for (int e = threadIdx.x; e < HDV_; e += 256) {
    int hh = e / DV_, d = e % DV_;
    vt[((size_t)hh * S_ + s) * DV_ + d] = (bf16_t)kvrow[hh * (DN_ + DV_) + DN_ + d];
  }
}

// ---------------------------------------------------------------- flash attention (causal)
// grid (S/128, H); block 256 = 8 waves; wave owns 16 query rows, full DV=128
__global__ __launch_bounds__(256) void k_attn(const bf16_t* __restrict__ qf,
                                              const bf16_t* __restrict__ kf,
                                              const bf16_t* __restrict__ vt,
                                              bf16_t* __restrict__ o) {
  __shared__ __align__(16) bf16_t Ks[32][DQ_];   // 12 KB
  __shared__ __align__(16) bf16_t VsT[DV_][32];  //  8 KB (transposed)
  __shared__ __align__(16) bf16_t Ps[8][16][32]; //  8 KB (wave-private P staging)
  const int h  = blockIdx.y;
  const int qb = blockIdx.x * 128;
  const int tid = threadIdx.x, wave = tid >> 5, lane = tid & 31;
  const int half_ = lane >> 4, l16 = lane & 15;
  const int qr0 = qb + wave * 16;
  const bf16_t* Q  = qf + (size_t)h * S_ * DQ_;
  const bf16_t* Kp = kf + (size_t)h * S_ * DQ_;
  const bf16_t* Vp = vt + (size_t)h * S_ * DV_;

  // preload Q fragments (6 x 16x32 bf16 A-frags, kept in VGPRs)
  V16 qa[6];
  {
    const bf16_t* qrow = Q + (size_t)(qr0 + l16) * DQ_;
    const int o0 = half_ ? 8 : 0, o1 = half_ ? 24 : 16;
#pragma unroll
    for (int f = 0; f < 6; ++f) {
      qa[f].u[0] = *(const uint4*)&qrow[f * 32 + o0];
      qa[f].u[1] = *(const uint4*)&qrow[f * 32 + o1];
    }
  }

  v8f oacc[8] = {};
  float mst[8], lst[8];
#pragma unroll
  for (int p = 0; p < 8; ++p) { mst[p] = -1e30f; lst[p] = 0.f; }
  const float scale = 0.07216878f; // 1/sqrt(192)

  for (int kt = 0; kt < qb + 128; kt += 32) {
#if HAVE_TDM
    // K tile 32x192 via TDM (one instruction, wave 0 only)
    if (wave == 0)
      tdm_load_tile_2d(Kp + (size_t)kt * DQ_, (unsigned)(uintptr_t)&Ks[0][0],
                       (unsigned)DQ_, (unsigned)S_, (unsigned)DQ_, 32u,
                       (unsigned long long)DQ_);
#else
#pragma unroll
    for (int r = 0; r < 3; ++r) {
      int u = r * 256 + tid;
      int row = u / 24, c8 = (u % 24) * 8;
      *(uint4*)&Ks[row][c8] = *(const uint4*)&Kp[(size_t)(kt + row) * DQ_ + c8];
    }
#endif
    // stage V tile transposed: 32x128 bf16, 2 uint4 per thread
#pragma unroll
    for (int r = 0; r < 2; ++r) {
      int u = r * 256 + tid;
      int row = u >> 4, c8 = (u & 15) * 8;
      U4B t; t.u = *(const uint4*)&Vp[(size_t)(kt + row) * DV_ + c8];
#pragma unroll
      for (int j = 0; j < 8; ++j) VsT[c8 + j][row] = t.e[j];
    }
#if HAVE_TDM
    if (wave == 0) __builtin_amdgcn_s_wait_tensorcnt(0);
#endif
    __syncthreads();

    if (kt <= qr0 + 15) { // tile intersects this wave's causal region (wave-uniform)
      // S = Q K^T : two 16-key subtiles, 6 K-steps of 32 over DQ
      v8f sc[2] = {};
#pragma unroll
      for (int f = 0; f < 6; ++f) {
#pragma unroll
        for (int jj = 0; jj < 2; ++jj) {
          V16 b;
          int key = jj * 16 + l16;
          int kb  = f * 32 + half_ * 8;
          b.u[0] = *(const uint4*)&Ks[key][kb];
          b.u[1] = *(const uint4*)&Ks[key][kb + 16];
          sc[jj] = wmma_bf16(qa[f].v, b.v, sc[jj]);
        }
      }
      // scale + causal mask + online softmax (rows live across 16-lane groups)
#pragma unroll
      for (int p = 0; p < 8; ++p) {
        int qrow = qr0 + p + 8 * half_;
        float s0 = sc[0][p] * scale, s1 = sc[1][p] * scale;
        if (kt + l16 > qrow)      s0 = -1e30f;
        if (kt + 16 + l16 > qrow) s1 = -1e30f;
        float v = fmaxf(s0, s1);
#pragma unroll
        for (int m = 1; m <= 8; m <<= 1) v = fmaxf(v, __shfl_xor(v, m, 32));
        float mnew  = fmaxf(mst[p], v);
        float alpha = __expf(mst[p] - mnew);
        float p0 = (s0 <= -1e29f) ? 0.f : __expf(s0 - mnew);
        float p1 = (s1 <= -1e29f) ? 0.f : __expf(s1 - mnew);
        float rs = p0 + p1;
#pragma unroll
        for (int m = 1; m <= 8; m <<= 1) rs += __shfl_xor(rs, m, 32);
        mst[p] = mnew;
        lst[p] = lst[p] * alpha + rs;
#pragma unroll
        for (int nt = 0; nt < 8; ++nt) oacc[nt][p] *= alpha;
        sc[0][p] = p0; sc[1][p] = p1;
      }
      // C-layout -> A-layout via wave-private LDS
#pragma unroll
      for (int jj = 0; jj < 2; ++jj)
#pragma unroll
        for (int p = 0; p < 8; ++p)
          Ps[wave][p + 8 * half_][jj * 16 + l16] = (bf16_t)sc[jj][p];
      asm volatile("s_wait_dscnt 0" ::: "memory"); // wave-local LDS RAW
      V16 pa;
      {
        const int o0 = half_ ? 8 : 0, o1 = half_ ? 24 : 16;
        pa.u[0] = *(const uint4*)&Ps[wave][l16][o0];
        pa.u[1] = *(const uint4*)&Ps[wave][l16][o1];
      }
      // O += P V : 8 N-tiles of 16 over DV
      const int kb0 = half_ * 8;
#pragma unroll
      for (int nt = 0; nt < 8; ++nt) {
        V16 b;
        int n = nt * 16 + l16;
        b.u[0] = *(const uint4*)&VsT[n][kb0];
        b.u[1] = *(const uint4*)&VsT[n][kb0 + 16];
        oacc[nt] = wmma_bf16(pa.v, b.v, oacc[nt]);
      }
    }
    __syncthreads();
  }
  // epilogue: normalize and store bf16 to o[S][H*DV]
#pragma unroll
  for (int nt = 0; nt < 8; ++nt) {
#pragma unroll
    for (int p = 0; p < 8; ++p) {
      int qrow = qr0 + p + 8 * half_;
      int d    = nt * 16 + l16;
      float val = oacc[nt][p] / lst[p];
      o[(size_t)qrow * HDV_ + h * DV_ + d] = (bf16_t)val;
    }
  }
}

// ---------------------------------------------------------------- host orchestration
extern "C" void kernel_launch(void* const* d_in, const int* in_sizes, int n_in,
                              void* d_out, int out_size, void* d_ws, size_t ws_size,
                              hipStream_t stream) {
  (void)in_sizes; (void)n_in; (void)out_size; (void)ws_size;
  const float* x     = (const float*)d_in[0];
  const float* w_q_a = (const float*)d_in[1];
  const float* q_ln  = (const float*)d_in[2];
  const float* w_q_b = (const float*)d_in[3];
  const float* w_kva = (const float*)d_in[4];
  const float* kv_ln = (const float*)d_in[5];
  const float* w_kvb = (const float*)d_in[6];
  const float* w_out = (const float*)d_in[7];
  float* out = (float*)d_out;

  char* ws = (char*)d_ws;
  size_t off = 0;
  auto alloc = [&](size_t bytes) -> void* {
    void* p = ws + off;
    off = (off + bytes + 255) & ~((size_t)255);
    return p;
  };

  bf16_t* x_bf    = (bf16_t*)alloc((size_t)S_ * E_ * 2);
  bf16_t* wqa_bf  = (bf16_t*)alloc((size_t)E_ * QL_ * 2);
  bf16_t* wqb_bf  = (bf16_t*)alloc((size_t)QL_ * HDQ_ * 2);
  bf16_t* wkva_bf = (bf16_t*)alloc((size_t)E_ * (KL_ + DR_) * 2);
  bf16_t* wkvb_bf = (bf16_t*)alloc((size_t)KL_ * HDNV_ * 2);
  bf16_t* wout_bf = (bf16_t*)alloc((size_t)HDV_ * E_ * 2);
  float*  qa_f    = (float*)alloc((size_t)S_ * QL_ * 4);
  bf16_t* qan_bf  = (bf16_t*)alloc((size_t)S_ * QL_ * 2);
  float*  q_f     = (float*)alloc((size_t)S_ * HDQ_ * 4);
  float*  ckv_f   = (float*)alloc((size_t)S_ * (KL_ + DR_) * 4);
  bf16_t* ckvn_bf = (bf16_t*)alloc((size_t)S_ * KL_ * 2);
  float*  kv_f    = (float*)alloc((size_t)S_ * HDNV_ * 4);
  bf16_t* qfb     = (bf16_t*)alloc((size_t)H_ * S_ * DQ_ * 2);
  bf16_t* kfb     = (bf16_t*)alloc((size_t)H_ * S_ * DQ_ * 2);
  bf16_t* vtb     = (bf16_t*)alloc((size_t)H_ * S_ * DV_ * 2);
  bf16_t* ob      = (bf16_t*)alloc((size_t)S_ * HDV_ * 2);

  auto cvt = [&](const float* src, bf16_t* dst, int n) {
    k_f32_to_bf16<<<dim3(256), dim3(256), 0, stream>>>(src, dst, n);
  };
  cvt(x,     x_bf,    S_ * E_);
  cvt(w_q_a, wqa_bf,  E_ * QL_);
  cvt(w_q_b, wqb_bf,  QL_ * HDQ_);
  cvt(w_kva, wkva_bf, E_ * (KL_ + DR_));
  cvt(w_kvb, wkvb_bf, KL_ * HDNV_);
  cvt(w_out, wout_bf, HDV_ * E_);

  // q_a = x @ w_q_a ; rmsnorm
  k_gemm<4><<<dim3(S_ / 128, QL_ / 128), 256, 0, stream>>>(x_bf, wqa_bf, qa_f, S_, QL_, E_);
  k_rmsnorm<<<dim3(S_), 256, 0, stream>>>(qa_f, q_ln, qan_bf, QL_, QL_);
  // q = q_a_n @ w_q_b
  k_gemm<4><<<dim3(S_ / 128, HDQ_ / 128), 256, 0, stream>>>(qan_bf, wqb_bf, q_f, S_, HDQ_, QL_);
  // ckv = x @ w_kv_a (N=576 -> BN=64 variant) ; rmsnorm of first KL cols
  k_gemm<2><<<dim3(S_ / 128, (KL_ + DR_) / 64), 256, 0, stream>>>(x_bf, wkva_bf, ckv_f, S_, KL_ + DR_, E_);
  k_rmsnorm<<<dim3(S_), 256, 0, stream>>>(ckv_f, kv_ln, ckvn_bf, KL_, KL_ + DR_);
  // kv = c_kv_n @ w_kv_b
  k_gemm<4><<<dim3(S_ / 128, HDNV_ / 128), 256, 0, stream>>>(ckvn_bf, wkvb_bf, kv_f, S_, HDNV_, KL_);
  // RoPE + repack to head-major bf16
  k_rope_pack<<<dim3(S_), 256, 0, stream>>>(q_f, kv_f, ckv_f, qfb, kfb, vtb);
  // causal flash attention
  k_attn<<<dim3(S_ / 128, H_), 256, 0, stream>>>(qfb, kfb, vtb, ob);
  // out = o @ w_out
  k_gemm<4><<<dim3(S_ / 128, E_ / 128), 256, 0, stream>>>(ob, wout_bf, out, S_, E_, HDV_);
}
